// MoELayer_71021579207358
// MI455X (gfx1250) — matrix operand order repaired and checked by reading, not compiled, not run on hardware
//
#include <hip/hip_runtime.h>
#include <cmath>

// ---------------- problem constants (match reference) ----------------
#define BB 2
#define SS 2048
#define DD 1024
#define EE 8
#define HH 4096
#define TT (BB * SS)        // 4096 tokens
#define TM 32               // tokens per FFN block (2 x 16-row WMMA subtiles)
#define HC 128              // H-chunk width (8 waves * 16)
#define NTILES (TT / TM)    // 128

typedef __attribute__((ext_vector_type(16))) __bf16 v16bf;
typedef __attribute__((ext_vector_type(8)))  __bf16 v8bf;
typedef __attribute__((ext_vector_type(8)))  float  v8f;
typedef __attribute__((ext_vector_type(4)))  unsigned int u32x4;
typedef __attribute__((ext_vector_type(8)))  int  i32x8;
typedef __attribute__((ext_vector_type(4)))  int  i32x4;

// ====================================================================
// Kernel 1: gating — one wave32 per token.
// ====================================================================
__global__ void moe_gate_kernel(const float* __restrict__ x,
                                const float* __restrict__ gw,
                                const float* __restrict__ gb,
                                float* __restrict__ probs,
                                int* __restrict__ counts,
                                int* __restrict__ lists) {
    const int lane = threadIdx.x & 31;
    const int wave = threadIdx.x >> 5;
    const int t = blockIdx.x * (blockDim.x >> 5) + wave;
    if (t >= TT) return;

    float acc[EE];
#pragma unroll
    for (int e = 0; e < EE; ++e) acc[e] = 0.f;

    const float* xr = x + (size_t)t * DD;
    for (int c = lane; c < DD; c += 32) {
        const float xv = xr[c];
#pragma unroll
        for (int e = 0; e < EE; ++e) acc[e] += xv * gw[c * EE + e];
    }
#pragma unroll
    for (int e = 0; e < EE; ++e) {
#pragma unroll
        for (int off = 16; off > 0; off >>= 1)
            acc[e] += __shfl_xor(acc[e], off, 32);
    }

    if (lane == 0) {
        float lg[EE];
        float mx = -3.402823466e+38f;
        int bi = 0;
#pragma unroll
        for (int e = 0; e < EE; ++e) {
            lg[e] = acc[e] + gb[e];
            if (lg[e] > mx) { mx = lg[e]; bi = e; }   // first-max like argmax
        }
        float sum = 0.f;
#pragma unroll
        for (int e = 0; e < EE; ++e) { lg[e] = expf(lg[e] - mx); sum += lg[e]; }
        const float inv = 1.f / sum;
#pragma unroll
        for (int e = 0; e < EE; ++e) probs[(size_t)t * EE + e] = lg[e] * inv;

        const int slot = atomicAdd(&counts[bi], 1);
        lists[bi * TT + slot] = t;
    }
}

// ====================================================================
// Kernel 2: aux loss = E * sum_{s,e} ( mean_b probs[b,s,e] )^2
// ====================================================================
__global__ void moe_aux_kernel(const float* __restrict__ probs,
                               float* __restrict__ out_aux) {
    __shared__ float red[256];
    float acc = 0.f;
    for (int i = threadIdx.x; i < SS * EE; i += blockDim.x) {
        const int s = i / EE;
        const int e = i - s * EE;
        const float m = 0.5f * (probs[(size_t)(0 * SS + s) * EE + e] +
                                probs[(size_t)(1 * SS + s) * EE + e]);
        acc += m * m;
    }
    red[threadIdx.x] = acc;
    __syncthreads();
    for (int off = 128; off > 0; off >>= 1) {
        if (threadIdx.x < off) red[threadIdx.x] += red[threadIdx.x + off];
        __syncthreads();
    }
    if (threadIdx.x == 0) out_aux[0] = red[0] * (float)EE;
}

// ====================================================================
// WMMA fragment helpers (contiguous k = kh*16 + i, same order for A and B)
// ====================================================================
// A (16-bit, 16x32): lane-half selects K sub-range kh*16..+15 (contiguous)
__device__ __forceinline__ v16bf load_a_frag(const __bf16* base, int row,
                                             int stride, int k0, int kh) {
    const v8bf lo = *(const v8bf*)&base[row * stride + k0 + kh * 16];
    const v8bf hi = *(const v8bf*)&base[row * stride + k0 + kh * 16 + 8];
    v16bf a;
#pragma unroll
    for (int i = 0; i < 8; ++i) { a[i] = lo[i]; a[8 + i] = hi[i]; }
    return a;
}
// B (16-bit, 32x16): lane column fixed, k = k0 + kh*16 + i (fp32 -> bf16)
__device__ __forceinline__ v16bf load_b_frag_f32(const float* p, int ld) {
    v16bf b;
#pragma unroll
    for (int i = 0; i < 16; ++i) b[i] = (__bf16)p[(size_t)i * ld];
    return b;
}

// ====================================================================
// TDM gather: 16 routed token rows (DD fp32 each) -> LDS, via Tensor
// Data Mover descriptor (CDNA5 ISA, gather mode, 16-bit indices).
// Tracked on TENSORcnt. This toolchain exposes the 6-arg builtin:
// (u32x4 g0, i32x8 g1, i32x4 g2, i32x4 g3, i32x8 extra, i32 cpol).
// ====================================================================
__device__ __forceinline__ void tdm_gather16_rows(const float* gbase,
                                                  unsigned int lds_off,
                                                  const int* tok, int first) {
    const unsigned long long ga = (unsigned long long)(uintptr_t)gbase;
    u32x4 g0; i32x8 g1; i32x4 g2, g3;
    // group0: count=1 | gather_mode(bit31), 16-bit indices (bit30=0)
    g0[0] = 1u | (1u << 31);
    g0[1] = lds_off;                                   // lds_addr (bytes)
    g0[2] = (unsigned int)ga;                          // global_addr[31:0]
    g0[3] = (unsigned int)((ga >> 32) & 0x01FFFFFFu)   // global_addr[56:32]
            | (2u << 30);                              // type = 2 ("image")
    // group1: data_size=4B(code 2); tensor 1024 x 4096; tile 1024 x 16 rows
    g1[0] = (int)(2u << 16);                           // data_size
    g1[1] = (int)(((unsigned)DD & 0xFFFFu) << 16);     // tensor_dim0[15:0]
    g1[2] = (int)(((unsigned)DD >> 16) | (((unsigned)TT & 0xFFFFu) << 16));
    g1[3] = (int)(((unsigned)TT >> 16) | ((unsigned)DD << 16)); // tile_dim0=DD
    g1[4] = 16;                                        // tile_dim1 = 16 indices
    g1[5] = DD;                                        // tensor_dim0_stride lo
    g1[6] = 0;
    g1[7] = 0;
    // groups 2/3: 16 packed 16-bit row indices (routed token ids)
#pragma unroll
    for (int j = 0; j < 4; ++j)
        g2[j] = (tok[first + 2 * j] & 0xFFFF) | (tok[first + 2 * j + 1] << 16);
#pragma unroll
    for (int j = 0; j < 4; ++j)
        g3[j] = (tok[first + 8 + 2 * j] & 0xFFFF) | (tok[first + 9 + 2 * j] << 16);

    i32x8 g4;                                          // unused extra group
#pragma unroll
    for (int j = 0; j < 8; ++j) g4[j] = 0;

#if __has_builtin(__builtin_amdgcn_tensor_load_to_lds)
    __builtin_amdgcn_tensor_load_to_lds(g0, g1, g2, g3, g4, 0);
#else
    asm volatile("tensor_load_to_lds %0, %1, %2, %3"
                 :: "s"(g0), "s"(g1), "s"(g2), "s"(g3) : "memory");
#endif
#if __has_builtin(__builtin_amdgcn_s_wait_tensorcnt)
    __builtin_amdgcn_s_wait_tensorcnt(0);
#else
    asm volatile("s_wait_tensorcnt 0x0" ::: "memory");
#endif
}

// ====================================================================
// Kernel 3: expert FFN, bf16 WMMA fp32-accumulate, 32-token tiles.
// grid = EE * NTILES blocks of 256 threads (8 waves).
// ====================================================================
__global__ __launch_bounds__(256, 1) void moe_ffn_kernel(
    const float* __restrict__ x,
    const float* __restrict__ w1, const float* __restrict__ b1,
    const float* __restrict__ w2, const float* __restrict__ b2,
    const int* __restrict__ counts, const int* __restrict__ lists,
    float* __restrict__ out) {

    const int e    = blockIdx.x / NTILES;
    const int tile = blockIdx.x - e * NTILES;
    const int n_e  = counts[e];
    const int base = tile * TM;
    if (base >= n_e) return;                 // block-uniform early exit
    int rows = n_e - base;
    if (rows > TM) rows = TM;

    __shared__ __bf16 Xs[TM * DD];           // 64 KB: 32 x 1024 bf16 A-tile
    __shared__ __bf16 Hs[TM * HC];           //  8 KB: 32 x 128 gelu chunk
    __shared__ float  Xf[16 * DD];           // 64 KB: TDM fp32 staging
    __shared__ int    tokS[TM];

    const int tid = threadIdx.x;
    if (tid < TM) {
        const int src = base + (tid < rows ? tid : 0);   // pad with valid token
        tokS[tid] = lists[e * TT + src];
    }
    __syncthreads();

    const int lane = tid & 31;
    const int wv   = tid >> 5;               // wave 0..7
    const int ln   = lane & 15;
    const int kh   = lane >> 4;

    // ---- stage X tile: TDM gathers 16 routed rows at a time into Xf,
    //      then all waves convert fp32 -> bf16 into Xs ----
    const unsigned int xf_off = (unsigned int)(uintptr_t)(&Xf[0]);
#pragma unroll
    for (int sub = 0; sub < 2; ++sub) {
        if (wv == 0) tdm_gather16_rows(x, xf_off, tokS, sub * 16);
        __syncthreads();                     // TDM data visible to block
        for (int idx = tid; idx < 16 * DD; idx += 256)
            Xs[(sub * 16 + (idx >> 10)) * DD + (idx & (DD - 1))] = (__bf16)Xf[idx];
        __syncthreads();                     // Xf reusable / Xs ready
    }

    const float* w1e = w1 + (size_t)e * DD * HH;
    const float* w2e = w2 + (size_t)e * HH * DD;

    v8f zero;
#pragma unroll
    for (int i = 0; i < 8; ++i) zero[i] = 0.f;

    v8f Y0[8], Y1[8];                        // 32 x (wv*128..+127) fp32 acc
#pragma unroll
    for (int nt = 0; nt < 8; ++nt) { Y0[nt] = zero; Y1[nt] = zero; }

    for (int ch = 0; ch < HH / HC; ++ch) {
        const int hbase = ch * HC;

        if (ch + 1 < HH / HC)                // hint next weight chunk to L2
            __builtin_prefetch(w1e + (size_t)(kh * 16) * HH +
                               (hbase + HC + wv * 16 + ln), 0, 1);

        // ---- GEMM1: acc{0,1}[16x16] = X(32xDD) x W1[:, hbase+wv*16..+15]
        //      one B fragment feeds both M-subtiles (2 WMMAs / 16 loads) ----
        v8f acc0 = zero, acc1 = zero;
        const int nH = hbase + wv * 16 + ln;
        for (int ks = 0; ks < DD / 32; ++ks) {
            const int k0 = ks * 32;
            const v16bf a0 = load_a_frag(Xs, ln,      DD, k0, kh);
            const v16bf a1 = load_a_frag(Xs, 16 + ln, DD, k0, kh);
            const v16bf b  = load_b_frag_f32(w1e + (size_t)(k0 + kh * 16) * HH + nH, HH);
            acc0 = __builtin_amdgcn_wmma_f32_16x16x32_bf16(
                       false, a0, false, b, (short)0, acc0, false, false);
            acc1 = __builtin_amdgcn_wmma_f32_16x16x32_bf16(
                       false, a1, false, b, (short)0, acc1, false, false);
        }

        __syncthreads();                     // previous chunk's Hs reads done

        // ---- bias + exact-erf GELU -> Hs (bf16) ----
        const float bias1 = b1[(size_t)e * HH + hbase + wv * 16 + ln];
#pragma unroll
        for (int r = 0; r < 8; ++r) {
            float v = acc0[r] + bias1;
            v = 0.5f * v * (1.f + erff(v * 0.70710678118654752f));
            Hs[(r + 8 * kh) * HC + wv * 16 + ln] = (__bf16)v;
            float u = acc1[r] + bias1;
            u = 0.5f * u * (1.f + erff(u * 0.70710678118654752f));
            Hs[(16 + r + 8 * kh) * HC + wv * 16 + ln] = (__bf16)u;
        }
        __syncthreads();

        // ---- GEMM2: Y += Hs(32x128) x W2[hbase..+127, wv*128..+127]
        //      k-step outermost keeps only 2 A fragments live ----
        for (int ks = 0; ks < 4; ++ks) {
            const int k0 = ks * 32;
            const v16bf ah0 = load_a_frag(Hs, ln,      HC, k0, kh);
            const v16bf ah1 = load_a_frag(Hs, 16 + ln, HC, k0, kh);
#pragma unroll
            for (int nt = 0; nt < 8; ++nt) {
                const int col = wv * 128 + nt * 16 + ln;
                const v16bf b = load_b_frag_f32(
                    w2e + (size_t)(hbase + k0 + kh * 16) * DD + col, DD);
                Y0[nt] = __builtin_amdgcn_wmma_f32_16x16x32_bf16(
                             false, ah0, false, b, (short)0, Y0[nt], false, false);
                Y1[nt] = __builtin_amdgcn_wmma_f32_16x16x32_bf16(
                             false, ah1, false, b, (short)0, Y1[nt], false, false);
            }
        }
    }

    // ---- epilogue: + b2, scatter valid rows to routed tokens ----
#pragma unroll
    for (int nt = 0; nt < 8; ++nt) {
        const int col = wv * 128 + nt * 16 + ln;
        const float bias2 = b2[(size_t)e * DD + col];
#pragma unroll
        for (int r = 0; r < 8; ++r) {
            const int m0 = r + 8 * kh;       // C/D layout: row = vgpr + 8*lanehalf
            const int m1 = 16 + m0;
            if (m0 < rows)
                out[(size_t)tokS[m0] * DD + col] = Y0[nt][r] + bias2;
            if (m1 < rows)
                out[(size_t)tokS[m1] * DD + col] = Y1[nt][r] + bias2;
        }
    }
}

// ====================================================================
// launcher
// ====================================================================
extern "C" void kernel_launch(void* const* d_in, const int* in_sizes, int n_in,
                              void* d_out, int out_size, void* d_ws, size_t ws_size,
                              hipStream_t stream) {
    const float* x  = (const float*)d_in[0];
    const float* gw = (const float*)d_in[1];
    const float* gb = (const float*)d_in[2];
    const float* w1 = (const float*)d_in[3];
    const float* b1 = (const float*)d_in[4];
    const float* w2 = (const float*)d_in[5];
    const float* b2 = (const float*)d_in[6];
    float* out = (float*)d_out;

    // workspace layout: probs[T*E] f32 | counts[32] i32 | lists[E*T] i32
    float* probs  = (float*)d_ws;
    int*   counts = (int*)((char*)d_ws + (size_t)TT * EE * sizeof(float));
    int*   lists  = counts + 32;

    (void)hipMemsetAsync(counts, 0, 32 * sizeof(int), stream);

    moe_gate_kernel<<<TT / 8, 256, 0, stream>>>(x, gw, gb, probs, counts, lists);
    moe_aux_kernel<<<1, 256, 0, stream>>>(probs, out + (size_t)TT * DD);
    moe_ffn_kernel<<<EE * NTILES, 256, 0, stream>>>(x, w1, b1, w2, b2,
                                                    counts, lists, out);
}
